// StructuralSelfAttention_60756607369326
// MI455X (gfx1250) — matrix operand order repaired and checked by reading, not compiled, not run on hardware
//
#include <hip/hip_runtime.h>
#include <hip/hip_bf16.h>
#include <math.h>

// Problem constants (from reference): B=8, E=2048, D=512, H=8, HD=64, NS=6
#define E_ 2048
#define D_ 512
#define H_ 8
#define HD_ 64
#define NS_ 6
#define SCALE_ 0.125f   // 64^-0.5

typedef __attribute__((ext_vector_type(16))) __bf16   v16bf;
typedef __attribute__((ext_vector_type(2)))  __bf16   v2bf;
typedef __attribute__((ext_vector_type(8)))  float    v8f;
typedef __attribute__((ext_vector_type(4)))  unsigned uint4v;
typedef __attribute__((ext_vector_type(2)))  unsigned uint2v;
typedef __attribute__((ext_vector_type(4)))  float    float4v;
typedef __attribute__((ext_vector_type(2)))  float    float2v;
typedef int v4i_gv __attribute__((vector_size(16)));   // GCC-vector int4 (builtin param type)

union FragA { v16bf v; uint4v q[2]; };           // 32 bytes
union U4S  { uint4v u; unsigned short s[8]; };

__device__ __forceinline__ unsigned short f2bf(float f) {
    unsigned u = __builtin_bit_cast(unsigned, f);
    u += 0x7FFFu + ((u >> 16) & 1u);             // round-to-nearest-even
    return (unsigned short)(u >> 16);
}

// packed f32x2 -> bf16x2 (single v_cvt_pk_bf16_f32 when available)
__device__ __forceinline__ unsigned pk2bf(float a, float b) {
#if __has_builtin(__builtin_amdgcn_cvt_pk_bf16_f32)
    v2bf r = __builtin_amdgcn_cvt_pk_bf16_f32(a, b);
    return __builtin_bit_cast(unsigned, r);
#else
    return (unsigned)f2bf(a) | ((unsigned)f2bf(b) << 16);
#endif
}

__device__ __forceinline__ v8f wmma_bf16(const FragA& a, const FragA& b, v8f c) {
    return __builtin_amdgcn_wmma_f32_16x16x32_bf16(false, a.v, false, b.v,
                                                   (short)0, c, false, false);
}

// ---- CDNA5 async global->LDS DMA path (ASYNCcnt), guarded so compile is safe
#if __has_builtin(__builtin_amdgcn_global_load_async_to_lds_b128)
#define HAS_ASYNC 1
#else
#define HAS_ASYNC 0
#endif

#if HAS_ASYNC
__device__ __forceinline__ void cp16_async(const unsigned short* g, unsigned short* l) {
    __builtin_amdgcn_global_load_async_to_lds_b128(
        (__attribute__((address_space(1))) v4i_gv*)g,
        (__attribute__((address_space(3))) v4i_gv*)l, 0, 0);
}
__device__ __forceinline__ void wait_async0() {
#if __has_builtin(__builtin_amdgcn_s_wait_asynccnt)
    __builtin_amdgcn_s_wait_asynccnt(0);
#else
    asm volatile("s_wait_asynccnt 0x0" ::: "memory");
#endif
}
#endif

// ---------------------------------------------------------------------------
// bf16 WMMA GEMM: C[M,N] = act(A[M,K] @ W + bias), W pre-transposed bf16 [N,K].
// Block tile 128x64, 8 waves (4x2), wave tile 32x32 (2x2 WMMA), BK=32.
// Double-buffered LDS; async global->LDS staging when available.
// MODE: 0 -> f32 row-major, 1 -> bf16 row-major, 2 -> bf16 [B,H,E,HD] layout
// ---------------------------------------------------------------------------
template<int MODE, int GELU>
__global__ __launch_bounds__(256) void gemm_k(const unsigned short* __restrict__ A,
                                              const unsigned short* __restrict__ Wt,
                                              const float* __restrict__ bias,
                                              void* __restrict__ outp,
                                              int M, int N, int K) {
    __shared__ alignas(16) unsigned short As[2][128][32];  // A tile, row-major
    __shared__ alignas(16) unsigned short Ws[2][64][40];   // W tile, n-major, padded

    const int tid = threadIdx.x;
    const int lane = tid & 31, wave = tid >> 5;
    const int wm = wave >> 1, wn = wave & 1;
    const int mrow = lane & 15, hi = lane >> 4;
    const unsigned m0 = blockIdx.y * 128u, n0 = blockIdx.x * 64u;

    // per-thread staging coordinates (16B chunks)
    const int am0 = tid >> 2,         ak0 = (tid & 3) * 8;
    const int am1 = (tid + 256) >> 2, ak1 = (tid & 3) * 8;
    const int wr0 = tid >> 2,         wk0 = (tid & 3) * 8;

    v8f acc[2][2];
#pragma unroll
    for (int a = 0; a < 2; a++)
#pragma unroll
        for (int b = 0; b < 2; b++)
#pragma unroll
            for (int r = 0; r < 8; r++) acc[a][b][r] = 0.f;

    uint4v ra0, ra1, rw0;   // fallback staging registers
    auto issue_tile = [&](int k0, int bsel) {
        const unsigned short* pa0 = A + (unsigned)(m0 + am0) * K + (unsigned)(k0 + ak0);
        const unsigned short* pa1 = A + (unsigned)(m0 + am1) * K + (unsigned)(k0 + ak1);
        const unsigned short* pw0 = Wt + (unsigned)(n0 + wr0) * K + (unsigned)(k0 + wk0);
#if HAS_ASYNC
        cp16_async(pa0, &As[bsel][am0][ak0]);
        cp16_async(pa1, &As[bsel][am1][ak1]);
        cp16_async(pw0, &Ws[bsel][wr0][wk0]);
#else
        (void)bsel;
        ra0 = *(const uint4v*)pa0;
        ra1 = *(const uint4v*)pa1;
        rw0 = *(const uint4v*)pw0;
#endif
    };
    auto commit_tile = [&](int bsel) {
#if !HAS_ASYNC
        *(uint4v*)&As[bsel][am0][ak0] = ra0;
        *(uint4v*)&As[bsel][am1][ak1] = ra1;
        *(uint4v*)&Ws[bsel][wr0][wk0] = rw0;
#else
        (void)bsel;
#endif
    };

    issue_tile(0, 0);
    commit_tile(0);
    int cur = 0;
    for (int k0 = 0; k0 < K; k0 += 32) {
#if HAS_ASYNC
        wait_async0();
#endif
        __syncthreads();

        FragA af[2], bfr[2];
#pragma unroll
        for (int t = 0; t < 2; t++) {
            int r = wm * 32 + t * 16 + mrow;
            af[t].q[0] = *(const uint4v*)&As[cur][r][8 * hi];       // K = 8hi..+7
            af[t].q[1] = *(const uint4v*)&As[cur][r][16 + 8 * hi];  // K = 16+8hi..+7
            int cc = wn * 32 + t * 16 + mrow;
            bfr[t].q[0] = *(const uint4v*)&Ws[cur][cc][8 * hi];     // K = 8hi..+7
            bfr[t].q[1] = *(const uint4v*)&Ws[cur][cc][16 + 8 * hi];
        }

        const bool more = (k0 + 32) < K;
        if (more) issue_tile(k0 + 32, cur ^ 1);
        if (k0 + 64 < K) {
            __builtin_prefetch(A + (unsigned)(m0 + am0) * K + (unsigned)(k0 + 64 + ak0), 0, 1);
            __builtin_prefetch(Wt + (unsigned)(n0 + wr0) * K + (unsigned)(k0 + 64 + wk0), 0, 1);
        }

#pragma unroll
        for (int a = 0; a < 2; a++)
#pragma unroll
            for (int b = 0; b < 2; b++)
                acc[a][b] = wmma_bf16(af[a], bfr[b], acc[a][b]);

        if (more) { commit_tile(cur ^ 1); cur ^= 1; }
    }

    // ---- epilogue ----
#pragma unroll
    for (int tm = 0; tm < 2; tm++)
#pragma unroll
        for (int tn = 0; tn < 2; tn++)
#pragma unroll
            for (int r = 0; r < 8; r++) {
                unsigned row = m0 + wm * 32 + tm * 16 + r + 8 * hi;
                unsigned col = n0 + wn * 32 + tn * 16 + mrow;
                float v = acc[tm][tn][r] + bias[col];
                if (GELU) v = 0.5f * v * (1.f + erff(v * 0.70710678f));
                if (MODE == 0) {
                    ((float*)outp)[row * (unsigned)N + col] = v;
                } else if (MODE == 1) {
                    ((unsigned short*)outp)[row * (unsigned)N + col] = f2bf(v);
                } else { // [B,H,E,HD] bf16 head layout
                    unsigned bb = row >> 11, e = row & (E_ - 1);
                    unsigned hh = col >> 6,  hd = col & (HD_ - 1);
                    ((unsigned short*)outp)[(((bb * H_ + hh) * E_) + e) * HD_ + hd] = f2bf(v);
                }
            }
}

// ---------------------------------------------------------------------------
// Flash attention with structural bias + key padding mask.
// Grid: B*H*(E/64) blocks, 128 threads (4 waves), each wave owns 16 queries.
// ---------------------------------------------------------------------------
__global__ __launch_bounds__(128) void attn_k(const unsigned short* __restrict__ Qb,
                                              const unsigned short* __restrict__ Kb,
                                              const unsigned short* __restrict__ Vb,
                                              const int* __restrict__ rel,
                                              const unsigned char* __restrict__ mask,
                                              const float* __restrict__ tbl,
                                              unsigned short* __restrict__ Ob) {
    __shared__ alignas(16) unsigned short Vs[64][72];      // V transposed [hd][k], padded
    __shared__ alignas(16) unsigned short Ps[4][16][64];   // per-wave P tile [q][k]
    __shared__ float tbl_l[NS_ * H_];
    __shared__ unsigned char mk[64];

    const int tid = threadIdx.x, lane = tid & 31, wave = tid >> 5;
    const int mrow = lane & 15, hi = lane >> 4;
    const int qb = blockIdx.x & 31;
    const int h  = (blockIdx.x >> 5) & (H_ - 1);
    const int b  = blockIdx.x >> 8;
    const size_t bh = (size_t)(b * H_ + h);
    const int q0 = qb * 64 + wave * 16;

    const unsigned short* Qp = Qb + bh * (E_ * HD_);   // uniform SGPR bases,
    const unsigned short* Kp = Kb + bh * (E_ * HD_);   // 32-bit per-lane offsets
    const unsigned short* Vp = Vb + bh * (E_ * HD_);
    const int* relp = rel + (size_t)b * E_ * E_;
    const unsigned char* mp = mask + (size_t)b * E_;
    unsigned short* Obp = Ob + (size_t)b * E_ * D_ + h * HD_;

    if (tid < NS_ * H_) tbl_l[tid] = tbl[tid];

    // Q A-fragments (loop invariant)
    FragA aq[2];
    {
        const unsigned qoff = (unsigned)(q0 + mrow) * HD_;
#pragma unroll
        for (int c = 0; c < 2; c++) {
            aq[c].q[0] = *(const uint4v*)(Qp + qoff + c * 32 + 8 * hi);
            aq[c].q[1] = *(const uint4v*)(Qp + qoff + c * 32 + 16 + 8 * hi);
        }
    }

    v8f oacc[4];
#pragma unroll
    for (int t = 0; t < 4; t++)
#pragma unroll
        for (int r = 0; r < 8; r++) oacc[t][r] = 0.f;
    float mprev[8], lsum[8];
#pragma unroll
    for (int r = 0; r < 8; r++) { mprev[r] = -3.0e38f; lsum[r] = 0.f; }

    for (int k0 = 0; k0 < E_; k0 += 64) {
        __syncthreads();
        // stage V block transposed: Vs[hd][k]
#pragma unroll
        for (int i = 0; i < 4; i++) {
            int c = tid + 128 * i;
            int kr = c >> 3, col = (c & 7) * 8;
            U4S u;
            u.u = *(const uint4v*)(Vp + (unsigned)(k0 + kr) * HD_ + col);
#pragma unroll
            for (int j = 0; j < 8; j++) Vs[col + j][kr] = u.s[j];
        }
        if (tid < 64) mk[tid] = mp[k0 + tid];
        __syncthreads();

        // ---- S = Q K^T * scale + struct bias, with mask ----
        float sv[4][8];
#pragma unroll
        for (int t = 0; t < 4; t++) {
            const unsigned koff = (unsigned)(k0 + t * 16 + mrow) * HD_;
            FragA b0, b1;
            b0.q[0] = *(const uint4v*)(Kp + koff + 8 * hi);
            b0.q[1] = *(const uint4v*)(Kp + koff + 16 + 8 * hi);
            b1.q[0] = *(const uint4v*)(Kp + koff + 32 + 8 * hi);
            b1.q[1] = *(const uint4v*)(Kp + koff + 32 + 16 + 8 * hi);
            v8f s = {0.f, 0.f, 0.f, 0.f, 0.f, 0.f, 0.f, 0.f};
            s = wmma_bf16(aq[0], b0, s);
            s = wmma_bf16(aq[1], b1, s);
            const int kcol = k0 + t * 16 + mrow;
            const bool km = mk[t * 16 + mrow] != 0;
#pragma unroll
            for (int r = 0; r < 8; r++) {
                unsigned roff = (unsigned)(q0 + r + 8 * hi) * E_ + kcol;
                int ridx = relp[roff];
                float x = s[r] * SCALE_ + tbl_l[ridx * H_ + h];
                sv[t][r] = km ? -1.0e30f : x;
            }
        }

        // ---- online softmax (rows live in 16-lane halves) ----
#pragma unroll
        for (int r = 0; r < 8; r++) {
            float mx = fmaxf(fmaxf(sv[0][r], sv[1][r]), fmaxf(sv[2][r], sv[3][r]));
            mx = fmaxf(mx, __shfl_xor(mx, 1));
            mx = fmaxf(mx, __shfl_xor(mx, 2));
            mx = fmaxf(mx, __shfl_xor(mx, 4));
            mx = fmaxf(mx, __shfl_xor(mx, 8));
            float mnew = fmaxf(mprev[r], mx);
            float corr = __expf(mprev[r] - mnew);
            float rs = 0.f;
#pragma unroll
            for (int t = 0; t < 4; t++) {
                float p = __expf(sv[t][r] - mnew);
                sv[t][r] = p;
                rs += p;
            }
            rs += __shfl_xor(rs, 1);
            rs += __shfl_xor(rs, 2);
            rs += __shfl_xor(rs, 4);
            rs += __shfl_xor(rs, 8);
            lsum[r] = lsum[r] * corr + rs;
            mprev[r] = mnew;
#pragma unroll
            for (int t = 0; t < 4; t++) oacc[t][r] = oacc[t][r] * corr;
        }

        // ---- P to LDS (C-layout write), reload as A-layout ----
#pragma unroll
        for (int t = 0; t < 4; t++)
#pragma unroll
            for (int r = 0; r < 8; r++)
                Ps[wave][r + 8 * hi][t * 16 + mrow] = f2bf(sv[t][r]);
        __syncthreads();

        // ---- O += P @ V ----
#pragma unroll
        for (int c = 0; c < 2; c++) {
            FragA ap;
            ap.q[0] = *(const uint4v*)&Ps[wave][mrow][c * 32 + 8 * hi];
            ap.q[1] = *(const uint4v*)&Ps[wave][mrow][c * 32 + 16 + 8 * hi];
#pragma unroll
            for (int t = 0; t < 4; t++) {
                FragA bv;
                bv.q[0] = *(const uint4v*)&Vs[t * 16 + mrow][c * 32 + 8 * hi];
                bv.q[1] = *(const uint4v*)&Vs[t * 16 + mrow][c * 32 + 16 + 8 * hi];
                oacc[t] = wmma_bf16(ap, bv, oacc[t]);
            }
        }
    }

    // ---- normalize and write O as bf16 [B,E,D] (heads concatenated) ----
#pragma unroll
    for (int t = 0; t < 4; t++)
#pragma unroll
        for (int r = 0; r < 8; r++) {
            unsigned qrow = q0 + r + 8 * hi;
            float ov = oacc[t][r] / lsum[r];
            Obp[qrow * D_ + t * 16 + mrow] = f2bf(ov);
        }
}

// ---------------------------------------------------------------------------
// out = LayerNorm(x + res); optionally also write bf16 copy for next GEMM.
// ---------------------------------------------------------------------------
template<int WB>
__global__ __launch_bounds__(256) void add_ln_k(const float* __restrict__ x,
                                                const float* __restrict__ res,
                                                const float* __restrict__ g,
                                                const float* __restrict__ be,
                                                float* __restrict__ out,
                                                unsigned short* __restrict__ outh) {
    __shared__ float ssum[8], ssq[8], stats[2];
    const int tid = threadIdx.x, lane = tid & 31, wave = tid >> 5;
    const size_t base = (size_t)blockIdx.x * D_;
    float v0 = x[base + tid] + res[base + tid];
    float v1 = x[base + tid + 256] + res[base + tid + 256];
    float s = v0 + v1, q = v0 * v0 + v1 * v1;
#pragma unroll
    for (int o = 16; o > 0; o >>= 1) {
        s += __shfl_down(s, o);
        q += __shfl_down(q, o);
    }
    if (lane == 0) { ssum[wave] = s; ssq[wave] = q; }
    __syncthreads();
    if (tid == 0) {
        float S = 0.f, Q = 0.f;
        for (int i = 0; i < 8; i++) { S += ssum[i]; Q += ssq[i]; }
        float mean = S / (float)D_;
        float var  = Q / (float)D_ - mean * mean;
        stats[0] = mean;
        stats[1] = rsqrtf(var + 1e-5f);
    }
    __syncthreads();
    float mean = stats[0], rs = stats[1];
    float y0 = (v0 - mean) * rs * g[tid] + be[tid];
    float y1 = (v1 - mean) * rs * g[tid + 256] + be[tid + 256];
    out[base + tid] = y0;
    out[base + tid + 256] = y1;
    if (WB) {
        outh[base + tid] = f2bf(y0);
        outh[base + tid + 256] = f2bf(y1);
    }
}

// ---------------------------------------------------------------------------
// One-shot preconversions (per launch; tiny vs 23.3 TB/s HBM):
// f32 -> bf16 elementwise (x), and f32 [K,N] -> bf16 [N,K] transpose (weights).
// ---------------------------------------------------------------------------
__global__ __launch_bounds__(256) void conv_k(const float* __restrict__ in,
                                              unsigned* __restrict__ out) {
    unsigned i = blockIdx.x * 256u + threadIdx.x;
    float2v f = *(const float2v*)(in + 2u * i);
    out[i] = pk2bf(f.x, f.y);
}

__global__ __launch_bounds__(256) void convT_k(const float* __restrict__ in,
                                               unsigned short* __restrict__ out,
                                               int K, int N) {
    __shared__ float t[32][33];
    const int tx = threadIdx.x & 31, ty = threadIdx.x >> 5;
    const unsigned k0 = blockIdx.y * 32u, n0 = blockIdx.x * 32u;
#pragma unroll
    for (int i = 0; i < 4; i++)
        t[ty + 8 * i][tx] = in[(k0 + ty + 8 * i) * (unsigned)N + n0 + tx];
    __syncthreads();
#pragma unroll
    for (int i = 0; i < 4; i++)
        out[(n0 + ty + 8 * i) * (unsigned)K + k0 + tx] = f2bf(t[tx][ty + 8 * i]);
}

// ---------------------------------------------------------------------------
extern "C" void kernel_launch(void* const* d_in, const int* in_sizes, int n_in,
                              void* d_out, int out_size, void* d_ws, size_t ws_size,
                              hipStream_t stream) {
    const float* x   = (const float*)d_in[0];
    const int*   srl = (const int*)d_in[1];
    const unsigned char* kpm = (const unsigned char*)d_in[2];
    const float* wq = (const float*)d_in[3];  const float* bq  = (const float*)d_in[4];
    const float* wk = (const float*)d_in[5];  const float* bk  = (const float*)d_in[6];
    const float* wv = (const float*)d_in[7];  const float* bvv = (const float*)d_in[8];
    const float* wo = (const float*)d_in[9];  const float* bo  = (const float*)d_in[10];
    const float* tbl = (const float*)d_in[11];
    const float* g1 = (const float*)d_in[12]; const float* be1 = (const float*)d_in[13];
    const float* w1 = (const float*)d_in[14]; const float* b1f = (const float*)d_in[15];
    const float* w2 = (const float*)d_in[16]; const float* b2f = (const float*)d_in[17];
    const float* g2 = (const float*)d_in[18]; const float* be2 = (const float*)d_in[19];

    const int M = 8 * E_;  // 16384 rows
    char* ws = (char*)d_ws;
    unsigned short* xb  = (unsigned short*)(ws);                        // 16 MB bf16 [M,D]
    unsigned short* Qb  = (unsigned short*)(ws + ((size_t)16  << 20));  // 16 MB [B,H,E,HD]
    unsigned short* Kb  = (unsigned short*)(ws + ((size_t)32  << 20));
    unsigned short* Vb  = (unsigned short*)(ws + ((size_t)48  << 20));
    unsigned short* Ob  = (unsigned short*)(ws + ((size_t)64  << 20));  // 16 MB [B,E,D]
    float* attno        = (float*)(ws + ((size_t)80  << 20));           // 32 MB
    float* x1f          = (float*)(ws + ((size_t)112 << 20));           // 32 MB
    unsigned short* x1h = (unsigned short*)(ws + ((size_t)144 << 20));  // 16 MB
    unsigned short* h1  = (unsigned short*)(ws + ((size_t)160 << 20));  // 64 MB [M,4D]
    float* ffo          = (float*)(ws + ((size_t)224 << 20));           // 32 MB
    unsigned short* wqT = (unsigned short*)(ws + ((size_t)256 << 20));          // [D,D]
    unsigned short* wkT = wqT + (size_t)D_ * D_;
    unsigned short* wvT = wkT + (size_t)D_ * D_;
    unsigned short* woT = wvT + (size_t)D_ * D_;
    unsigned short* w1T = woT + (size_t)D_ * D_;          // [4D, D]
    unsigned short* w2T = w1T + (size_t)4 * D_ * D_;      // [D, 4D]

    dim3 blk(256);
    // preconvert x and weights to bf16 (weights transposed to [N,K])
    conv_k<<<dim3((M * D_ / 2) / 256), blk, 0, stream>>>(x, (unsigned*)xb);
    convT_k<<<dim3(D_ / 32, D_ / 32), blk, 0, stream>>>(wq, wqT, D_, D_);
    convT_k<<<dim3(D_ / 32, D_ / 32), blk, 0, stream>>>(wk, wkT, D_, D_);
    convT_k<<<dim3(D_ / 32, D_ / 32), blk, 0, stream>>>(wv, wvT, D_, D_);
    convT_k<<<dim3(D_ / 32, D_ / 32), blk, 0, stream>>>(wo, woT, D_, D_);
    convT_k<<<dim3(4 * D_ / 32, D_ / 32), blk, 0, stream>>>(w1, w1T, D_, 4 * D_);
    convT_k<<<dim3(D_ / 32, 4 * D_ / 32), blk, 0, stream>>>(w2, w2T, 4 * D_, D_);

    dim3 g512(D_ / 64, M / 128);
    dim3 g2048((4 * D_) / 64, M / 128);

    // QKV projections -> bf16 head layout
    gemm_k<2, 0><<<g512, blk, 0, stream>>>(xb, wqT, bq,  Qb, M, D_, D_);
    gemm_k<2, 0><<<g512, blk, 0, stream>>>(xb, wkT, bk,  Kb, M, D_, D_);
    gemm_k<2, 0><<<g512, blk, 0, stream>>>(xb, wvT, bvv, Vb, M, D_, D_);

    // Flash attention with structural bias
    attn_k<<<dim3(8 * H_ * (E_ / 64)), dim3(128), 0, stream>>>(Qb, Kb, Vb, srl, kpm, tbl, Ob);

    // Output projection (f32), residual + LN1 (f32 + bf16 copy)
    gemm_k<0, 0><<<g512, blk, 0, stream>>>(Ob, woT, bo, attno, M, D_, D_);
    add_ln_k<1><<<dim3(M), blk, 0, stream>>>(x, attno, g1, be1, x1f, x1h);

    // FFN: gelu(x1 @ w1 + b1) @ w2 + b2, residual + LN2
    gemm_k<1, 1><<<g2048, blk, 0, stream>>>(x1h, w1T, b1f, h1, M, 4 * D_, D_);
    gemm_k<0, 0><<<g512, blk, 0, stream>>>(h1, w2T, b2f, ffo, M, D_, 4 * D_);   // K = 4*D (fixed)
    add_ln_k<0><<<dim3(M), blk, 0, stream>>>(x1f, ffo, g2, be2, (float*)d_out, nullptr);
}
